// HybridModel_5720896438538
// MI455X (gfx1250) — matrix operand order repaired and checked by reading, not compiled, not run on hardware
//
#include <hip/hip_runtime.h>
#include <hip/hip_bf16.h>
#include <math.h>

// Model dims (compile-time constants from the reference)
#define Bz 8
#define Nz 256
#define Dz 48
#define Vz 20
#define Lz 2
#define Sz 6

typedef _Float16 half16 __attribute__((ext_vector_type(16)));
typedef float    float8 __attribute__((ext_vector_type(8)));

__device__ __forceinline__ float gelu_exact(float z) {
    return 0.5f * z * (1.0f + erff(z * 0.70710678118654752f));
}

// ---------------------------------------------------------------------------
// x[b,n,:] = embed[ids[b,n],:] + pos[n,:]
// ---------------------------------------------------------------------------
__global__ __launch_bounds__(64) void embed_kernel(
    const int* __restrict__ ids, const float* __restrict__ emb,
    const float* __restrict__ pos, float* __restrict__ x) {
    int bn = blockIdx.x;            // b*N + n
    int d  = threadIdx.x;
    if (d < Dz)
        x[bn * Dz + d] = emb[ids[bn] * Dz + d] + pos[(bn & (Nz - 1)) * Dz + d];
}

// ---------------------------------------------------------------------------
// Pre-swizzle pair_w1[96,192-split] into 24 fp16 WMMA B-fragments (K padded
// 48->64, zero fill). nt 0..5 -> u outputs (rows 0..47 of w1), nt 6..11 -> v
// outputs (rows 48..95). f = nt*2 + kc, kc in {0,1} covering K 0..31 / 32..63.
// B layout: lane = Kb*16 + n, halfword t -> K = kc*32 + (t<8 ? Kb*8+t : 16+Kb*8+t-8).
// ---------------------------------------------------------------------------
__global__ __launch_bounds__(256) void w1frag_kernel(
    const float* __restrict__ w1, _Float16* __restrict__ w1f) {
    int idx = blockIdx.x * 256 + threadIdx.x;
    if (idx >= 24 * 512) return;
    int f  = idx >> 9;
    int r  = idx & 511;
    int ln = r >> 4, t = r & 15;
    int nt = f >> 1,  kc = f & 1;
    int n  = ln & 15, Kb = ln >> 4;
    int K  = kc * 32 + ((t < 8) ? (Kb * 8 + t) : (16 + Kb * 8 + (t - 8)));   // 0..63
    float val = 0.0f;
    if (K < 48) {
        int row = (nt < 6) ? K : (48 + K);
        int col = ((nt < 6) ? nt : (nt - 6)) * 16 + n;
        val = w1[row * 96 + col];
    }
    w1f[idx] = (_Float16)val;
}

// ---------------------------------------------------------------------------
// u/v projection on WMMA: [2048 x 48] @ [48 x 192].  One block stages 128
// tokens into LDS in A-fragment layout (K padded to 64); each of 8 waves owns
// one M-tile and runs 12 N-tiles x 2 K-steps = 24 WMMAs.
// ---------------------------------------------------------------------------
__global__ __launch_bounds__(256) void uv_kernel(
    const float* __restrict__ x, const _Float16* __restrict__ w1f,
    const float* __restrict__ b1, float* __restrict__ u, float* __restrict__ v) {
    __shared__ __align__(32) _Float16 abuf[128 * 64];   // 16KB
    const int tid  = threadIdx.x;
    const int lane = tid & 31;
    const int wave = tid >> 5;
    const int tok0 = blockIdx.x * 128;

    for (int idx = tid; idx < 128 * 64; idx += 256) {
        int tt = idx >> 6;                   // token within block
        int c  = idx & 63;                   // padded K index
        float val = (c < Dz) ? x[(tok0 + tt) * Dz + c] : 0.0f;
        int mt = tt >> 4, m = tt & 15;
        int kc = c >> 5,  k = c & 31;
        int h, t;
        if (k < 16) { h = k >> 3;        t = k & 7; }
        else        { h = (k - 16) >> 3; t = 8 + ((k - 16) & 7); }
        abuf[(mt * 2 + kc) * 512 + (h * 16 + m) * 16 + t] = (_Float16)val;
    }
    __syncthreads();

    const int mt  = wave;
    half16 a0 = *(const half16*)(abuf + (mt * 2 + 0) * 512 + lane * 16);
    half16 a1 = *(const half16*)(abuf + (mt * 2 + 1) * 512 + lane * 16);
    const int n_  = lane & 15;
    const int hh_ = lane >> 4;
    #pragma unroll
    for (int nt = 0; nt < 12; ++nt) {
        half16 bf0 = *(const half16*)(w1f + (nt * 2 + 0) * 512 + lane * 16);
        half16 bf1 = *(const half16*)(w1f + (nt * 2 + 1) * 512 + lane * 16);
        float8 acc = {};
        acc = __builtin_amdgcn_wmma_f32_16x16x32_f16(false, a0, false, bf0,
                                                     (short)0, acc, false, false);
        acc = __builtin_amdgcn_wmma_f32_16x16x32_f16(false, a1, false, bf1,
                                                     (short)0, acc, false, false);
        bool  isU  = (nt < 6);
        int   col  = (isU ? nt : (nt - 6)) * 16 + n_;
        float bias = isU ? b1[col] : 0.0f;
        float* dst = isU ? u : v;
        #pragma unroll
        for (int r = 0; r < 8; ++r) {
            int m   = r + 8 * hh_;
            int tok = tok0 + mt * 16 + m;
            dst[tok * 96 + col] = acc[r] + bias;
        }
    }
}

// ---------------------------------------------------------------------------
// Pre-swizzle pair_w2[96,48] (fp32) into 9 fp16 WMMA B-fragments.
// ---------------------------------------------------------------------------
__global__ __launch_bounds__(256) void w2frag_kernel(
    const float* __restrict__ w2, _Float16* __restrict__ w2f) {
    int idx = blockIdx.x * 256 + threadIdx.x;
    if (idx >= 9 * 512) return;
    int f  = idx >> 9;
    int r  = idx & 511;
    int ln = r >> 4, t = r & 15;
    int nt = f / 3,  kc = f - nt * 3;
    int n  = ln & 15, Kb = ln >> 4;
    int K  = kc * 32 + ((t < 8) ? (Kb * 8 + t) : (16 + Kb * 8 + (t - 8)));   // 0..95
    w2f[idx] = (_Float16)w2[K * Dz + nt * 16 + n];
}

// ---------------------------------------------------------------------------
// Pair module, one workgroup per (b, i):
//   h[j,c]  = GELU(u[b,i,c] + v[b,j,c])      (written swizzled-fp16 to LDS)
//   p[j,:]  = h[j,:] @ w2 + b2               (WMMA f32_16x16x32_f16, masked)
//   w[j]    = softmax_j( ||p[j,:]|| , mask ) ; out = sum_j w[j]*p[j,:]
// ---------------------------------------------------------------------------
__global__ __launch_bounds__(256) void pair_kernel(
    const float* __restrict__ u, const float* __restrict__ v,
    const _Float16* __restrict__ w2f, const float* __restrict__ b2,
    float* __restrict__ pout) {
    // 49152B buffer: first holds 48 A-fragments (fp16), then reused as p[256][48] f32.
    __shared__ __align__(32) unsigned char smem[Nz * 96 * 2];
    __shared__ float u_lds[96];
    __shared__ float red[Nz];
    __shared__ float scr[Nz];

    const int tid  = threadIdx.x;
    const int lane = tid & 31;
    const int wave = tid >> 5;
    const int bi   = blockIdx.x;          // b*N + i
    const int b    = bi >> 8;
    const int i    = bi & 255;

    if (tid < 96) u_lds[tid] = u[bi * 96 + tid];
    __syncthreads();

    // h -> LDS, directly in A-fragment layout (16-bit A 16x32 per ISA 7.12.2)
    _Float16* hbuf = (_Float16*)smem;
    const float* vbase = v + (size_t)b * Nz * 96;   // linear: j*96 + c == idx
    for (int idx = tid; idx < Nz * 96; idx += 256) {
        __builtin_prefetch(vbase + idx + 512, 0, 0);    // global_prefetch_b8
        float z = u_lds[idx % 96] + vbase[idx];
        float g = gelu_exact(z);
        int j = idx / 96;
        int c = idx - j * 96;
        int mt = j >> 4, m = j & 15;
        int kc = c >> 5, k = c & 31;
        int h, t;
        if (k < 16) { h = k >> 3;        t = k & 7; }
        else        { h = (k - 16) >> 3; t = 8 + ((k - 16) & 7); }
        hbuf[(mt * 3 + kc) * 512 + (h * 16 + m) * 16 + t] = (_Float16)g;
    }
    __syncthreads();

    // All 9 w2 B-fragments resident in VGPRs (72 VGPRs)
    half16 bf[9];
    #pragma unroll
    for (int f = 0; f < 9; ++f)
        bf[f] = *(const half16*)(w2f + f * 512 + lane * 16);

    // Each wave owns 2 M-tiles; p rows of tile mt overwrite exactly the LDS
    // bytes of tile mt's own A-fragments (read-before-write within the wave).
    float* pmat = (float*)smem;           // [256][48] f32
    const int n_  = lane & 15;
    const int hh_ = lane >> 4;
    #pragma unroll
    for (int mi = 0; mi < 2; ++mi) {
        int mt = wave * 2 + mi;
        float8 acc0 = {}, acc1 = {}, acc2 = {};
        #pragma unroll
        for (int kc = 0; kc < 3; ++kc) {
            half16 af = *(const half16*)(hbuf + (mt * 3 + kc) * 512 + lane * 16);
            acc0 = __builtin_amdgcn_wmma_f32_16x16x32_f16(false, af, false, bf[0 + kc],
                                                          (short)0, acc0, false, false);
            acc1 = __builtin_amdgcn_wmma_f32_16x16x32_f16(false, af, false, bf[3 + kc],
                                                          (short)0, acc1, false, false);
            acc2 = __builtin_amdgcn_wmma_f32_16x16x32_f16(false, af, false, bf[6 + kc],
                                                          (short)0, acc2, false, false);
        }
        #pragma unroll
        for (int r = 0; r < 8; ++r) {
            int m = r + 8 * hh_;          // C/D layout: VGPR r, lanes16-31 -> M+8
            int j = mt * 16 + m;
            float keep = (j > i) ? 0.0f : 1.0f;   // causal mask (j > i masked)
            pmat[j * Dz +  0 + n_] = keep * (acc0[r] + b2[ 0 + n_]);
            pmat[j * Dz + 16 + n_] = keep * (acc1[r] + b2[16 + n_]);
            pmat[j * Dz + 32 + n_] = keep * (acc2[r] + b2[32 + n_]);
        }
    }
    __syncthreads();

    // w[j] = ||p[j,:]||  (masked -> -1e9), then block softmax over j
    float sq = 0.0f;
    {
        const float* prow = pmat + tid * Dz;
        #pragma unroll 8
        for (int c = 0; c < Dz; ++c) sq = fmaf(prow[c], prow[c], sq);
    }
    float wj = (tid > i) ? -1e9f : sqrtf(sq);
    scr[tid] = wj;
    __syncthreads();
    for (int s = 128; s > 0; s >>= 1) {
        if (tid < s) scr[tid] = fmaxf(scr[tid], scr[tid + s]);
        __syncthreads();
    }
    float mx = scr[0];
    __syncthreads();
    float e = __expf(wj - mx);
    scr[tid] = e;
    __syncthreads();
    for (int s = 128; s > 0; s >>= 1) {
        if (tid < s) scr[tid] += scr[tid + s];
        __syncthreads();
    }
    float denom = scr[0];
    __syncthreads();
    red[tid] = e / denom;
    __syncthreads();

    // out[b,i,c] = sum_j w[j] * p[j,c]
    if (tid < Dz) {
        float a = 0.0f;
        for (int j = 0; j < Nz; ++j) a = fmaf(red[j], pmat[j * Dz + tid], a);
        pout[bi * Dz + tid] = a;
    }
}

// ---------------------------------------------------------------------------
// Triple module, one workgroup per (b, n): S=6 neighbors, 3D=144 -> 96 -> 48
// ---------------------------------------------------------------------------
__global__ __launch_bounds__(96) void triple_kernel(
    const float* __restrict__ x, const int* __restrict__ ji,
    const int* __restrict__ ki, const float* __restrict__ w1,
    const float* __restrict__ b1, const float* __restrict__ w2,
    const float* __restrict__ b2, float* __restrict__ tout) {
    __shared__ float cat[Sz][144];
    __shared__ float hid[Sz][96];
    int bn  = blockIdx.x;               // b*N + n
    int b   = bn >> 8;
    int tid = threadIdx.x;

    for (int idx = tid; idx < Sz * 144; idx += 96) {
        int s = idx / 144, d = idx - s * 144;
        float val;
        if (d < 48)      val = x[bn * Dz + d];
        else if (d < 96) val = x[(b * Nz + ji[bn * Sz + s]) * Dz + (d - 48)];
        else             val = x[(b * Nz + ki[bn * Sz + s]) * Dz + (d - 96)];
        cat[s][d] = val;
    }
    __syncthreads();
    {
        int o = tid;                    // 0..95
        #pragma unroll
        for (int s = 0; s < Sz; ++s) {
            float a = b1[o];
            for (int d = 0; d < 144; ++d) a = fmaf(cat[s][d], w1[d * 96 + o], a);
            hid[s][o] = gelu_exact(a);
        }
    }
    __syncthreads();
    if (tid < Dz) {
        float a = 0.0f;
        #pragma unroll
        for (int s = 0; s < Sz; ++s) {
            float acc = 0.0f;
            for (int o = 0; o < 96; ++o) acc = fmaf(hid[s][o], w2[o * Dz + tid], acc);
            a += acc;
        }
        tout[bn * Dz + tid] = a * (1.0f / 6.0f) + b2[tid];
    }
}

// ---------------------------------------------------------------------------
// Gate + residual mix + LayerNorm (n1), one workgroup per token
// ---------------------------------------------------------------------------
__global__ __launch_bounds__(64) void gate_ln_kernel(
    const float* __restrict__ p, const float* __restrict__ t,
    const float* __restrict__ gw, const float* __restrict__ gb,
    const float* __restrict__ ng, const float* __restrict__ nb,
    float* __restrict__ x) {
    __shared__ float cat[96];
    __shared__ float y[Dz];
    __shared__ float stats[2];
    int bn  = blockIdx.x;
    int tid = threadIdx.x;
    if (tid < Dz) { cat[tid] = p[bn * Dz + tid]; cat[Dz + tid] = t[bn * Dz + tid]; }
    __syncthreads();
    if (tid < Dz) {
        float a = gb[tid];
        for (int d = 0; d < 96; ++d) a = fmaf(cat[d], gw[d * Dz + tid], a);
        float g = 1.0f / (1.0f + __expf(-a));
        y[tid] = x[bn * Dz + tid] + g * cat[tid] + (1.0f - g) * cat[Dz + tid];
    }
    __syncthreads();
    if (tid == 0) {
        float mu = 0.0f;
        for (int d = 0; d < Dz; ++d) mu += y[d];
        mu *= (1.0f / Dz);
        float var = 0.0f;
        for (int d = 0; d < Dz; ++d) { float q = y[d] - mu; var = fmaf(q, q, var); }
        stats[0] = mu;
        stats[1] = rsqrtf(var * (1.0f / Dz) + 1e-5f);
    }
    __syncthreads();
    if (tid < Dz)
        x[bn * Dz + tid] = (y[tid] - stats[0]) * stats[1] * ng[tid] + nb[tid];
}

// ---------------------------------------------------------------------------
// FFN (48 -> 192 -> 48) + residual + LayerNorm (n2), one workgroup per token
// ---------------------------------------------------------------------------
__global__ __launch_bounds__(192) void ffn_kernel(
    const float* __restrict__ w1, const float* __restrict__ b1,
    const float* __restrict__ w2, const float* __restrict__ b2,
    const float* __restrict__ ng, const float* __restrict__ nb,
    float* __restrict__ x) {
    __shared__ float xr[Dz];
    __shared__ float hid[192];
    __shared__ float y[Dz];
    __shared__ float stats[2];
    int bn  = blockIdx.x;
    int tid = threadIdx.x;
    if (tid < Dz) xr[tid] = x[bn * Dz + tid];
    __syncthreads();
    {
        float a = b1[tid];
        #pragma unroll 8
        for (int d = 0; d < Dz; ++d) a = fmaf(xr[d], w1[d * 192 + tid], a);
        hid[tid] = gelu_exact(a);
    }
    __syncthreads();
    if (tid < Dz) {
        float a = b2[tid];
        for (int o = 0; o < 192; ++o) a = fmaf(hid[o], w2[o * Dz + tid], a);
        y[tid] = xr[tid] + a;
    }
    __syncthreads();
    if (tid == 0) {
        float mu = 0.0f;
        for (int d = 0; d < Dz; ++d) mu += y[d];
        mu *= (1.0f / Dz);
        float var = 0.0f;
        for (int d = 0; d < Dz; ++d) { float q = y[d] - mu; var = fmaf(q, q, var); }
        stats[0] = mu;
        stats[1] = rsqrtf(var * (1.0f / Dz) + 1e-5f);
    }
    __syncthreads();
    if (tid < Dz)
        x[bn * Dz + tid] = (y[tid] - stats[0]) * stats[1] * ng[tid] + nb[tid];
}

// ---------------------------------------------------------------------------
// Head: out[b,n,v] = x[b,n,:] @ head_w + head_b
// ---------------------------------------------------------------------------
__global__ __launch_bounds__(256) void head_kernel(
    const float* __restrict__ x, const float* __restrict__ hw,
    const float* __restrict__ hb, float* __restrict__ out) {
    int idx = blockIdx.x * 256 + threadIdx.x;
    if (idx >= Bz * Nz * Vz) return;
    int t = idx / Vz, vv = idx - t * Vz;
    float a = hb[vv];
    #pragma unroll 8
    for (int d = 0; d < Dz; ++d) a = fmaf(x[t * Dz + d], hw[d * Vz + vv], a);
    out[idx] = a;
}

// ---------------------------------------------------------------------------
extern "C" void kernel_launch(void* const* d_in, const int* in_sizes, int n_in,
                              void* d_out, int out_size, void* d_ws, size_t ws_size,
                              hipStream_t stream) {
    (void)in_sizes; (void)n_in; (void)out_size; (void)ws_size;
    const int*   ids     = (const int*)d_in[0];
    const int*   ji      = (const int*)d_in[1];
    const int*   ki      = (const int*)d_in[2];
    const float* emb     = (const float*)d_in[3];
    const float* pos     = (const float*)d_in[4];
    const float* pair_w1 = (const float*)d_in[5];
    const float* pair_b1 = (const float*)d_in[6];
    const float* pair_w2 = (const float*)d_in[7];
    const float* pair_b2 = (const float*)d_in[8];
    const float* tri_w1  = (const float*)d_in[9];
    const float* tri_b1  = (const float*)d_in[10];
    const float* tri_w2  = (const float*)d_in[11];
    const float* tri_b2  = (const float*)d_in[12];
    const float* gate_w  = (const float*)d_in[13];
    const float* gate_b  = (const float*)d_in[14];
    const float* n1g     = (const float*)d_in[15];
    const float* n1b     = (const float*)d_in[16];
    const float* ffn_w1  = (const float*)d_in[17];
    const float* ffn_b1  = (const float*)d_in[18];
    const float* ffn_w2  = (const float*)d_in[19];
    const float* ffn_b2  = (const float*)d_in[20];
    const float* n2g     = (const float*)d_in[21];
    const float* n2b     = (const float*)d_in[22];
    const float* head_w  = (const float*)d_in[23];
    const float* head_b  = (const float*)d_in[24];

    char* ws = (char*)d_ws;
    const int NT = Bz * Nz;                       // 2048 tokens
    size_t off = 0;
    float* x = (float*)(ws + off);   off += (size_t)NT * Dz * 4;       // 393216
    float* u = (float*)(ws + off);   off += (size_t)NT * 96 * 4;       // 786432
    float* v = (float*)(ws + off);   off += (size_t)NT * 96 * 4;
    float* p = (float*)(ws + off);   off += (size_t)NT * Dz * 4;
    float* t = (float*)(ws + off);   off += (size_t)NT * Dz * 4;
    _Float16* w2f = (_Float16*)(ws + off);  off += 16384;              // 9*512*2, padded
    _Float16* w1f = (_Float16*)(ws + off);  off += 32768;              // 24*512*2, padded

    embed_kernel<<<NT, 64, 0, stream>>>(ids, emb, pos, x);
    for (int l = 0; l < Lz; ++l) {
        w1frag_kernel<<<48, 256, 0, stream>>>(pair_w1 + l * 96 * 96, w1f);
        uv_kernel<<<NT / 128, 256, 0, stream>>>(x, w1f, pair_b1 + l * 96, u, v);
        w2frag_kernel<<<18, 256, 0, stream>>>(pair_w2 + l * 96 * Dz, w2f);
        pair_kernel<<<NT, 256, 0, stream>>>(u, v, w2f, pair_b2 + l * Dz, p);
        triple_kernel<<<NT, 96, 0, stream>>>(
            x, ji + l * NT * Sz, ki + l * NT * Sz,
            tri_w1 + l * 144 * 96, tri_b1 + l * 96,
            tri_w2 + l * 96 * Dz, tri_b2 + l * Dz, t);
        gate_ln_kernel<<<NT, 64, 0, stream>>>(
            p, t, gate_w + l * 96 * Dz, gate_b + l * Dz,
            n1g + l * Dz, n1b + l * Dz, x);
        ffn_kernel<<<NT, 192, 0, stream>>>(
            ffn_w1 + l * Dz * 192, ffn_b1 + l * 192,
            ffn_w2 + l * 192 * Dz, ffn_b2 + l * Dz,
            n2g + l * Dz, n2b + l * Dz, x);
    }
    head_kernel<<<(Bz * Nz * Vz + 255) / 256, 256, 0, stream>>>(
        x, head_w, head_b, (float*)d_out);
}
